// MFWRelativeSelfMultiheadAttn_42468636623452
// MI455X (gfx1250) — matrix operand order repaired and checked by reading, not compiled, not run on hardware
//
#include <hip/hip_runtime.h>
#include <hip/hip_bf16.h>

#define E_  1024
#define H_  16
#define D_  64
#define T_  1024
#define B_  4
#define TB_ (T_ * B_)   // 4096
#define E3_ (3 * E_)    // 3072

typedef __attribute__((ext_vector_type(16))) __bf16         v16bf;
typedef __attribute__((ext_vector_type(8)))  float          v8f;
typedef __attribute__((ext_vector_type(16))) unsigned short u16x16;
typedef __attribute__((ext_vector_type(8)))  unsigned short u16x8;
typedef __attribute__((vector_size(16)))     int            i32x4;
typedef __attribute__((address_space(1)))    i32x4*         gp_i32x4;
typedef __attribute__((address_space(3)))    i32x4*         lp_i32x4;

union U16V { u16x16 v; unsigned short s[16]; };

#if defined(__gfx1250__) && __has_builtin(__builtin_amdgcn_global_load_async_to_lds_b128)
#define HAVE_ASYNC 1
#else
#define HAVE_ASYNC 0
#endif

__device__ __forceinline__ unsigned short f2bfu(float f) {
  unsigned u = __builtin_bit_cast(unsigned, f);
  unsigned r = u + 0x7FFFu + ((u >> 16) & 1u);   // round-to-nearest-even
  return (unsigned short)(r >> 16);
}
__device__ __forceinline__ float bfu2f(unsigned short h) {
  unsigned u = ((unsigned)h) << 16;
  return __builtin_bit_cast(float, u);
}
__device__ __forceinline__ v16bf as_bf(u16x16 u) {
  return __builtin_bit_cast(v16bf, u);
}
__device__ __forceinline__ v8f wmma_bf16(v16bf a, v16bf b, v8f c) {
  return __builtin_amdgcn_wmma_f32_16x16x32_bf16(
      /*neg_a=*/false, a, /*neg_b=*/false, b,
      /*c_mod=*/(short)0, c, /*reuse_a=*/false, /*reuse_b=*/false);
}
// A fragment (16x32 bf16): row m = lane&15; elems 0..15 at K=k0+hi*16+e
// (two contiguous 16B loads).
__device__ __forceinline__ v16bf load_a_frag(const unsigned short* aRow, int k0,
                                             int hi) {
  u16x8 lo = *(const u16x8*)(aRow + k0 + hi * 16);
  u16x8 hh = *(const u16x8*)(aRow + k0 + hi * 16 + 8);
  return as_bf(__builtin_shufflevector(lo, hh, 0, 1, 2, 3, 4, 5, 6, 7, 8, 9, 10,
                                       11, 12, 13, 14, 15));
}

// ---------------------------------------------------------------------------
// W_fact = W * outer(rm, sm) + sum_{k<4} outer(r_k, s_k), emitted as bf16 bits
// ---------------------------------------------------------------------------
__global__ void factorize_kernel(const float* __restrict__ W,
                                 const float* __restrict__ rmB,
                                 const float* __restrict__ smB,
                                 const float* __restrict__ rB,
                                 const float* __restrict__ sB,
                                 const int* __restrict__ idxP,
                                 int rows, int cols,
                                 unsigned short* __restrict__ out) {
  const int idx = idxP[0];
  const float* rm = rmB + (size_t)idx * rows;
  const float* sm = smB + (size_t)idx * cols;
  const float* r  = rB + (size_t)idx * 4 * rows;
  const float* s  = sB + (size_t)idx * 4 * cols;
  const size_t n = (size_t)rows * cols;
  for (size_t ii = (size_t)blockIdx.x * blockDim.x + threadIdx.x; ii < n;
       ii += (size_t)gridDim.x * blockDim.x) {
    const int a = (int)(ii / cols), bc = (int)(ii % cols);
    float v = W[ii] * rm[a] * sm[bc];
#pragma unroll
    for (int k = 0; k < 4; ++k) v += r[(size_t)k * rows + a] * s[(size_t)k * cols + bc];
    out[ii] = f2bfu(v);
  }
}

__global__ void cvt_bf16_kernel(const float* __restrict__ in,
                                unsigned short* __restrict__ out, size_t n) {
  for (size_t i = (size_t)blockIdx.x * blockDim.x + threadIdx.x; i < n;
       i += (size_t)gridDim.x * blockDim.x)
    out[i] = f2bfu(in[i]);
}

// ---------------------------------------------------------------------------
// C[m,n] = sum_k A[m,k] * B[n,k] + bias[n]  (bf16 in, fp32 acc, bf16/fp32 out)
// Register-blocked: each wave owns a 32(M) x 64(N) tile -> 8 WMMA / k-step
// from 2 A-fragments x 4 B-fragments. 8 waves/block => 256 x 64 per block.
// Requires M % 256 == 0, N % 64 == 0.
// ---------------------------------------------------------------------------
__global__ __launch_bounds__(256) void gemm_nt_wmma(
    const unsigned short* __restrict__ A, const unsigned short* __restrict__ Bm,
    const float* __restrict__ bias, unsigned short* __restrict__ outBf,
    float* __restrict__ outF, int M, int N, int K) {
  const int lane = threadIdx.x & 31, lr = lane & 15, hi = lane >> 4;
  const int wv = threadIdx.x >> 5;
  const int m0 = (blockIdx.y * 8 + wv) * 32;
  const int n0 = blockIdx.x * 64;
  const unsigned short* aRow[2];
  const unsigned short* bRow[4];
#pragma unroll
  for (int mi = 0; mi < 2; ++mi)
    aRow[mi] = A + (size_t)(m0 + mi * 16 + lr) * K;
#pragma unroll
  for (int nt = 0; nt < 4; ++nt)
    bRow[nt] = Bm + (size_t)(n0 + nt * 16 + lr) * K + hi * 16;

  v8f acc[2][4];
#pragma unroll
  for (int mi = 0; mi < 2; ++mi)
#pragma unroll
    for (int nt = 0; nt < 4; ++nt)
#pragma unroll
      for (int r = 0; r < 8; ++r) acc[mi][nt][r] = 0.f;

  for (int k0 = 0; k0 < K; k0 += 32) {
    __builtin_prefetch(aRow[0] + k0 + 256, 0, 1);
    __builtin_prefetch(aRow[1] + k0 + 256, 0, 1);
    v16bf af[2], bf[4];
#pragma unroll
    for (int mi = 0; mi < 2; ++mi) af[mi] = load_a_frag(aRow[mi], k0, hi);
#pragma unroll
    for (int nt = 0; nt < 4; ++nt)
      bf[nt] = as_bf(*(const u16x16*)(bRow[nt] + k0));
#pragma unroll
    for (int mi = 0; mi < 2; ++mi)
#pragma unroll
      for (int nt = 0; nt < 4; ++nt)
        acc[mi][nt] = wmma_bf16(af[mi], bf[nt], acc[mi][nt]);
  }
#pragma unroll
  for (int mi = 0; mi < 2; ++mi)
#pragma unroll
    for (int nt = 0; nt < 4; ++nt) {
      const int col = n0 + nt * 16 + lr;
      const float bv = bias ? bias[col] : 0.f;
#pragma unroll
      for (int r = 0; r < 8; ++r) {
        const int row = m0 + mi * 16 + hi * 8 + r;
        const float v = acc[mi][nt][r] + bv;
        if (outF) outF[(size_t)row * N + col] = v;
        else      outBf[(size_t)row * N + col] = f2bfu(v);
      }
    }
}

// ---------------------------------------------------------------------------
// Fused relative attention with exact Transformer-XL rel_shift semantics.
// One wave per (16-query tile, b, h); flash-style online softmax, 32 keys/step.
// V tiles staged to LDS with gfx1250 async global->LDS copies when available.
// ---------------------------------------------------------------------------
__global__ __launch_bounds__(128) void attn_wmma(
    const unsigned short* __restrict__ qkv,   // [TB, 3E] bf16 (q|k|v)
    const unsigned short* __restrict__ rpos,  // [T, E]   bf16
    const float* __restrict__ r_w_bias,       // [H*D]
    const float* __restrict__ r_r_bias,       // [H*D]
    unsigned short* __restrict__ AO) {        // [TB, E]  bf16
  __shared__ float lM1[4][16 * 32];
  __shared__ float lM2[4][16 * 32];
  __shared__ float lP[4][16 * 32];
  __shared__ unsigned short lV[4][32 * 64];   // per-wave V tile (keys x dims)
  const int lane = threadIdx.x & 31, lr = lane & 15, hi = lane >> 4;
  const int wv = threadIdx.x >> 5;
  const int gw = blockIdx.x * 4 + wv;
  const int qt = gw & 63;            // 64 query tiles
  const int bh = gw >> 6;            // 0..63
  const int b = bh & (B_ - 1);
  const int h = bh >> 2;
  const int t0 = qt * 16;
  const float scale = 0.125f;        // 1/sqrt(D)

  // Persistent A fragments: Qw = q+r_w_bias, Qr = q+r_r_bias, Qs = shifted row
  v16bf aQw[2], aQr[2], aQs[2];
  {
    const int trow = t0 + lr;
    const int trowS = (trow + 1 < T_) ? (trow + 1) : (T_ - 1);
    const size_t qb = ((size_t)trow * B_ + b) * E3_ + h * D_;
    const size_t qbS = ((size_t)trowS * B_ + b) * E3_ + h * D_;
#pragma unroll
    for (int k = 0; k < 2; ++k) {
      U16V uw, ur, us;
#pragma unroll
      for (int e = 0; e < 16; ++e) {
        const int d = k * 32 + hi * 16 + e;
        const float q0 = bfu2f(qkv[qb + d]);
        const float qs = bfu2f(qkv[qbS + d]);
        uw.s[e] = f2bfu(q0 + r_w_bias[h * D_ + d]);
        ur.s[e] = f2bfu(q0 + r_r_bias[h * D_ + d]);
        us.s[e] = f2bfu(qs + r_r_bias[h * D_ + d]);
      }
      aQw[k] = as_bf(uw.v); aQr[k] = as_bf(ur.v); aQs[k] = as_bf(us.v);
    }
  }

  float mrun[8], lrun[8];
  v8f O[4];
#pragma unroll
  for (int r = 0; r < 8; ++r) { mrun[r] = -3.0e38f; lrun[r] = 0.f; }
#pragma unroll
  for (int nt = 0; nt < 4; ++nt)
#pragma unroll
    for (int r = 0; r < 8; ++r) O[nt][r] = 0.f;

  for (int s0 = 0; s0 < T_; s0 += 32) {
#if HAVE_ASYNC
    // Kick off async staging of this 32x64 V tile (one key row per lane,
    // 8 x 16B chunks) while we compute the scores for the same key block.
    {
      const unsigned short* vrow =
          qkv + ((size_t)(s0 + lane) * B_ + b) * E3_ + 2 * E_ + h * D_;
      unsigned short* ldst = &lV[wv][lane * 64];
#pragma unroll
      for (int c = 0; c < 8; ++c)
        __builtin_amdgcn_global_load_async_to_lds_b128(
            (gp_i32x4)(void*)(vrow + c * 8),
            (lp_i32x4)(void*)(ldst + c * 8), 0, 0);
    }
#endif
    float sc[2][8];
#pragma unroll
    for (int half = 0; half < 2; ++half) {
      const int ss = s0 + half * 16;
      const int d0 = t0 - ss;
      // ---- AC tile: (q + r_w_bias) . k -----------------------------------
      v8f accS = {0.f, 0.f, 0.f, 0.f, 0.f, 0.f, 0.f, 0.f};
      {
        const size_t kb =
            ((size_t)(ss + lr) * B_ + b) * E3_ + E_ + h * D_ + hi * 16;
#pragma unroll
        for (int k = 0; k < 2; ++k) {
          u16x16 bu = *(const u16x16*)(qkv + kb + k * 32);
          accS = wmma_bf16(aQw[k], as_bf(bu), accS);
        }
      }
      // ---- BD bands (rel_shift): M1 for s<=t, M2 for s>=t+2 --------------
      const bool needM1 = (ss <= t0 + 15);
      const bool needM2 = (ss + 15 >= t0 + 2);
      if (needM1) {
        const int base1 = T_ - 1 - d0 - 15;  // BD[t,s]=Qr_t.r[base1 + 15-i+j]
#pragma unroll
        for (int part = 0; part < 2; ++part) {
          v8f accM = {0.f, 0.f, 0.f, 0.f, 0.f, 0.f, 0.f, 0.f};
          const int u = base1 + part * 16 + lr;
          const unsigned short* rp = rpos + (size_t)u * E_ + h * D_ + hi * 16;
#pragma unroll
          for (int k = 0; k < 2; ++k) {
            U16V bu;
#pragma unroll
            for (int e = 0; e < 16; ++e) bu.s[e] = 0;
            if (u >= 0 && u < T_) bu.v = *(const u16x16*)(rp + k * 32);
            accM = wmma_bf16(aQr[k], as_bf(bu.v), accM);
          }
#pragma unroll
          for (int r = 0; r < 8; ++r)
            lM1[wv][(hi * 8 + r) * 32 + part * 16 + lr] = accM[r];
        }
      }
      if (needM2) {
        const int base2 = -d0 - 17;  // BD[t,s]=Qr_{t+1}.r[base2 + 15+j-i]
#pragma unroll
        for (int part = 0; part < 2; ++part) {
          v8f accM = {0.f, 0.f, 0.f, 0.f, 0.f, 0.f, 0.f, 0.f};
          const int u = base2 + part * 16 + lr;
          const unsigned short* rp = rpos + (size_t)u * E_ + h * D_ + hi * 16;
#pragma unroll
          for (int k = 0; k < 2; ++k) {
            U16V bu;
#pragma unroll
            for (int e = 0; e < 16; ++e) bu.s[e] = 0;
            if (u >= 0 && u < T_) bu.v = *(const u16x16*)(rp + k * 32);
            accM = wmma_bf16(aQs[k], as_bf(bu.v), accM);
          }
#pragma unroll
          for (int r = 0; r < 8; ++r)
            lM2[wv][(hi * 8 + r) * 32 + part * 16 + lr] = accM[r];
        }
      }
      asm volatile("s_wait_dscnt 0" ::: "memory");  // wave-local LDS bounce
      // ---- combine AC + shifted BD ---------------------------------------
#pragma unroll
      for (int r = 0; r < 8; ++r) {
        const int i = hi * 8 + r, j = lr;
        const int t = t0 + i, s = ss + j;
        float bd;
        if (s <= t)          bd = lM1[wv][i * 32 + (15 - i + j)];
        else if (s == t + 1) bd = 0.f;
        else                 bd = lM2[wv][i * 32 + (15 + j - i)];
        sc[half][r] = (accS[r] + bd) * scale;
      }
    }
    // ---- online softmax over this 32-key block ---------------------------
    float pr0[8], pr1[8];
#pragma unroll
    for (int r = 0; r < 8; ++r) {
      float tmax = fmaxf(sc[0][r], sc[1][r]);
#pragma unroll
      for (int off = 8; off >= 1; off >>= 1)
        tmax = fmaxf(tmax, __shfl_xor(tmax, off, 32));
      const float mnew = fmaxf(mrun[r], tmax);
      const float alpha = __expf(mrun[r] - mnew);
      const float p0 = __expf(sc[0][r] - mnew);
      const float p1 = __expf(sc[1][r] - mnew);
      float psum = p0 + p1;
#pragma unroll
      for (int off = 8; off >= 1; off >>= 1) psum += __shfl_xor(psum, off, 32);
      lrun[r] = lrun[r] * alpha + psum;
      mrun[r] = mnew;
#pragma unroll
      for (int nt = 0; nt < 4; ++nt) O[nt][r] *= alpha;
      pr0[r] = p0; pr1[r] = p1;
    }
#pragma unroll
    for (int r = 0; r < 8; ++r) {
      const int i = hi * 8 + r;
      lP[wv][i * 32 + lr] = pr0[r];
      lP[wv][i * 32 + 16 + lr] = pr1[r];
    }
    asm volatile("s_wait_dscnt 0" ::: "memory");
    // ---- P (C-layout -> A-layout via LDS) and P @ V ----------------------
    U16V pu;
#pragma unroll
    for (int e = 0; e < 16; ++e) {
      const int kk = hi * 16 + e;
      pu.s[e] = f2bfu(lP[wv][lr * 32 + kk]);
    }
    const v16bf pa = as_bf(pu.v);
#if HAVE_ASYNC
#if __has_builtin(__builtin_amdgcn_s_wait_asynccnt)
    __builtin_amdgcn_s_wait_asynccnt(0);
#else
    asm volatile("s_wait_asynccnt 0" ::: "memory");
#endif
    asm volatile("" ::: "memory");
#endif
#pragma unroll
    for (int nt = 0; nt < 4; ++nt) {
      U16V vu;
#pragma unroll
      for (int e = 0; e < 16; ++e) {
        const int kk = hi * 16 + e;  // key offset in block
#if HAVE_ASYNC
        vu.s[e] = lV[wv][kk * 64 + nt * 16 + lr];
#else
        vu.s[e] = qkv[((size_t)(s0 + kk) * B_ + b) * E3_ + 2 * E_ + h * D_ +
                      nt * 16 + lr];
#endif
      }
      O[nt] = wmma_bf16(pa, as_bf(vu.v), O[nt]);
    }
  }
  // ---- normalize and emit ------------------------------------------------
#pragma unroll
  for (int nt = 0; nt < 4; ++nt)
#pragma unroll
    for (int r = 0; r < 8; ++r) {
      const int i = hi * 8 + r;
      const float v = O[nt][r] / lrun[r];
      AO[((size_t)(t0 + i) * B_ + b) * E_ + h * D_ + nt * 16 + lr] = f2bfu(v);
    }
}

// ---------------------------------------------------------------------------
extern "C" void kernel_launch(void* const* d_in, const int* in_sizes, int n_in,
                              void* d_out, int out_size, void* d_ws,
                              size_t ws_size, hipStream_t stream) {
  const float* input = (const float*)d_in[0];
  const float* pos   = (const float*)d_in[1];
  const int*   idx   = (const int*)d_in[2];
  const float* in_w  = (const float*)d_in[3];
  const float* in_b  = (const float*)d_in[4];
  const float* out_w = (const float*)d_in[5];
  const float* out_b = (const float*)d_in[6];
  const float* pos_w = (const float*)d_in[7];
  const float* pos_b = (const float*)d_in[8];
  const float* r_i = (const float*)d_in[9],  *s_i = (const float*)d_in[10];
  const float* r_p = (const float*)d_in[11], *s_p = (const float*)d_in[12];
  const float* r_o = (const float*)d_in[13], *s_o = (const float*)d_in[14];
  const float* rm_i = (const float*)d_in[15], *sm_i = (const float*)d_in[16];
  const float* rm_p = (const float*)d_in[17], *sm_p = (const float*)d_in[18];
  const float* rm_o = (const float*)d_in[19], *sm_o = (const float*)d_in[20];
  const float* rwb = (const float*)d_in[21];
  const float* rrb = (const float*)d_in[22];

  // bf16 workspace layout (~57 MB total)
  unsigned short* ws  = (unsigned short*)d_ws;
  unsigned short* Wi  = ws;                              // [3072,1024]
  unsigned short* Wp  = Wi  + (size_t)E3_ * E_;          // [1024,1024]
  unsigned short* Wo  = Wp  + (size_t)E_ * E_;           // [1024,1024]
  unsigned short* Xb  = Wo  + (size_t)E_ * E_;           // [4096,1024]
  unsigned short* Pb  = Xb  + (size_t)TB_ * E_;          // [1024,1024]
  unsigned short* QKV = Pb  + (size_t)T_ * E_;           // [4096,3072]
  unsigned short* Rb  = QKV + (size_t)TB_ * E3_;         // [1024,1024]
  unsigned short* AO  = Rb  + (size_t)T_ * E_;           // [4096,1024]

  factorize_kernel<<<1024, 256, 0, stream>>>(in_w, rm_i, sm_i, r_i, s_i, idx,
                                             E3_, E_, Wi);
  factorize_kernel<<<512, 256, 0, stream>>>(pos_w, rm_p, sm_p, r_p, s_p, idx,
                                            E_, E_, Wp);
  factorize_kernel<<<512, 256, 0, stream>>>(out_w, rm_o, sm_o, r_o, s_o, idx,
                                            E_, E_, Wo);
  cvt_bf16_kernel<<<1024, 256, 0, stream>>>(input, Xb, (size_t)TB_ * E_);
  cvt_bf16_kernel<<<512, 256, 0, stream>>>(pos, Pb, (size_t)T_ * E_);

  // qkv = X @ Wi^T + in_bias   -> bf16 [4096, 3072]
  gemm_nt_wmma<<<dim3(E3_ / 64, TB_ / 256), 256, 0, stream>>>(
      Xb, Wi, in_b, QKV, nullptr, TB_, E3_, E_);
  // r = pos @ Wp^T + pos_bias  -> bf16 [1024, 1024]
  gemm_nt_wmma<<<dim3(E_ / 64, T_ / 256), 256, 0, stream>>>(
      Pb, Wp, pos_b, Rb, nullptr, T_, E_, E_);
  // fused relative attention -> bf16 [4096, 1024]
  attn_wmma<<<1024, 128, 0, stream>>>(QKV, Rb, rwb, rrb, AO);
  // out = AO @ Wo^T + out_bias -> fp32 d_out [4096, 1024]
  gemm_nt_wmma<<<dim3(E_ / 64, TB_ / 256), 256, 0, stream>>>(
      AO, Wo, out_b, nullptr, (float*)d_out, TB_, E_, E_);
}
